// ViT_15650860826823
// MI455X (gfx1250) — matrix operand order repaired and checked by reading, not compiled, not run on hardware
//
#include <hip/hip_runtime.h>
#include <hip/hip_bf16.h>
#include <stdint.h>

// ---------------- problem constants ----------------
#define B_    256
#define S_    50
#define D_    1024
#define NH_   2
#define HD_   512
#define DFF_  4096
#define L_    8
#define OUT_  10
#define M_    (B_ * S_)   // 12800 rows
#define MPAD_ 12816       // padded rows for q/k/v (attention tile over-read)

typedef __attribute__((ext_vector_type(16))) __bf16 v16bf;
typedef __attribute__((ext_vector_type(8)))  float  v8f;

__device__ __forceinline__ unsigned short f2bf(float f) {
  unsigned int u = __float_as_uint(f);
  u += 0x7FFFu + ((u >> 16) & 1u);          // round-to-nearest-even
  return (unsigned short)(u >> 16);
}
__device__ __forceinline__ float bf2f(unsigned short h) {
  return __uint_as_float(((unsigned int)h) << 16);
}
__device__ __forceinline__ unsigned int pack2bf(float a, float b) {
  return (unsigned int)f2bf(a) | ((unsigned int)f2bf(b) << 16);
}

// --- CDNA5 async global->LDS copy (ASYNCcnt-tracked, copy engine writes LDS) ---
// Low 32 bits of a generic pointer to LDS are the LDS byte address (aperture scheme).
__device__ __forceinline__ void async_load_b128(unsigned int lds_off,
                                                const void* gaddr) {
  asm volatile("global_load_async_to_lds_b128 %0, %1, off"
               :: "v"(lds_off), "v"((unsigned long long)(size_t)gaddr)
               : "memory");
}
__device__ __forceinline__ void wait_asynccnt0() {
  asm volatile("s_wait_asynccnt 0x0" ::: "memory");
}

// ---------------- patch embed + class token + pos emb ----------------
__global__ __launch_bounds__(256)
void embed_kernel(const float* __restrict__ X, const float* __restrict__ Wl,
                  const float* __restrict__ bl, const float* __restrict__ cls,
                  float* __restrict__ Out)
{
  const int row = blockIdx.x;            // 0..12799 = b*50 + s
  const int b = row / S_, s = row % S_;
  __shared__ float patch[16];
  if (s > 0 && threadIdx.x < 16) {
    int pidx = s - 1, py = pidx / 7, px = pidx % 7;
    int r = threadIdx.x >> 2, cc = threadIdx.x & 3;
    patch[threadIdx.x] = X[(size_t)b * 784 + (py * 4 + r) * 28 + (px * 4 + cc)];
  }
  __syncthreads();
  for (int d = threadIdx.x; d < D_; d += 256) {
    float val;
    if (s == 0) {
      val = cls[d];
    } else {
      float acc = bl[d];
#pragma unroll
      for (int f = 0; f < 16; ++f) acc += patch[f] * Wl[d * 16 + f];
      val = acc;
    }
    int je = d & ~1;
    float expo = (float)je / (float)D_;
    float ang = (float)s * powf(10000.0f, -expo);
    float pe = (d & 1) ? cosf(ang) : sinf(ang);
    Out[(size_t)row * D_ + d] = val + pe;
  }
}

// ---------------- LayerNorm -> bf16 ----------------
__global__ __launch_bounds__(128)
void ln_kernel(const float* __restrict__ X, const float* __restrict__ g,
               const float* __restrict__ bta, unsigned short* __restrict__ H)
{
  const int row = blockIdx.x;
  const int tid = threadIdx.x;
  const float* x = X + (size_t)row * D_;
  float v[8];
  float s = 0.f, ss = 0.f;
#pragma unroll
  for (int i = 0; i < 8; ++i) {
    float f = x[tid + i * 128];
    v[i] = f; s += f; ss += f * f;
  }
  __shared__ float red0[128];
  __shared__ float red1[128];
  red0[tid] = s; red1[tid] = ss;
  __syncthreads();
  for (int off = 64; off > 0; off >>= 1) {
    if (tid < off) { red0[tid] += red0[tid + off]; red1[tid] += red1[tid + off]; }
    __syncthreads();
  }
  float mean = red0[0] * (1.0f / (float)D_);
  float var  = red1[0] * (1.0f / (float)D_) - mean * mean;
  float inv  = rsqrtf(var + 1e-5f);
#pragma unroll
  for (int i = 0; i < 8; ++i) {
    int d = tid + i * 128;
    float hn = (v[i] - mean) * inv * g[d] + bta[d];
    H[(size_t)row * D_ + d] = f2bf(hn);
  }
}

// ---------------- bf16 WMMA GEMM: C[M,N] = A[M,K] @ W[N,K]^T + bias ----------------
// MODE 0: store bf16          (QKV)
// MODE 1: exact GELU, bf16    (MLP1)
// MODE 2: accumulate into f32 (MLP2 residual)
// A-tiles: async global->LDS (bf16 pass-through). W-tiles: f32->bf16 via VGPRs.
// Double-buffered LDS, one barrier per K-step.
#define TM 128
#define TN 128
#define TK 32
#define LDSK 40   // padded LDS stride (bf16 elems) -> conflict-free b128 reads

template<int MODE>
__global__ __launch_bounds__(256)
void gemm_bf16(const unsigned short* __restrict__ A, const float* __restrict__ W,
               const float* __restrict__ bias, unsigned short* __restrict__ Obf,
               float* __restrict__ Of, int N, int K)
{
  __shared__ __align__(16) unsigned short As[2][TM * LDSK];
  __shared__ __align__(16) unsigned short Bs[2][TN * LDSK];
  const int tid  = threadIdx.x;
  const int wave = tid >> 5, lane = tid & 31;
  const int m0 = blockIdx.y * TM, n0 = blockIdx.x * TN;
  const int wm = (wave >> 2) * 64;   // 2 waves in M
  const int wn = (wave & 3) * 32;    // 4 waves in N
  const int lr = lane & 15;
  const int kA = (lane < 16) ? 0 : 8;    // A frag: lanes>=16 hold K=8..15 / 24..31
  const int kB = (lane < 16) ? 0 : 8;    // B frag: lanes>=16 hold K=8..15 / 24..31

  v8f zero = {0.f, 0.f, 0.f, 0.f, 0.f, 0.f, 0.f, 0.f};
  v8f acc[4][2];
#pragma unroll
  for (int i = 0; i < 4; ++i)
#pragma unroll
    for (int j = 0; j < 2; ++j) acc[i][j] = zero;

  const int arow = tid >> 2;           // 0..63
  const int acol = (tid & 3) * 8;      // 0,8,16,24
  const int brow = tid >> 1;           // 0..127
  const int bcol = (tid & 1) * 16;     // 0,16

  const unsigned short* Ag0 = A + (size_t)(m0 + arow) * K + acol;
  const unsigned short* Ag1 = A + (size_t)(m0 + arow + 64) * K + acol;
  const float*          Wg  = W + (size_t)(n0 + brow) * K + bcol;
  unsigned int aL0[2], aL1[2];
#pragma unroll
  for (int bf = 0; bf < 2; ++bf) {
    aL0[bf] = (unsigned int)(size_t)&As[bf][arow * LDSK + acol];
    aL1[bf] = (unsigned int)(size_t)&As[bf][(arow + 64) * LDSK + acol];
  }

  // ---- prologue: stage tile 0 ----
  async_load_b128(aL0[0], Ag0);
  async_load_b128(aL1[0], Ag1);
  {
    const float4* ws = (const float4*)(Wg);
    float4 f0 = ws[0], f1 = ws[1], f2 = ws[2], f3 = ws[3];
    uint4 p0, p1;
    p0.x = pack2bf(f0.x, f0.y); p0.y = pack2bf(f0.z, f0.w);
    p0.z = pack2bf(f1.x, f1.y); p0.w = pack2bf(f1.z, f1.w);
    p1.x = pack2bf(f2.x, f2.y); p1.y = pack2bf(f2.z, f2.w);
    p1.z = pack2bf(f3.x, f3.y); p1.w = pack2bf(f3.z, f3.w);
    *(uint4*)&Bs[0][brow * LDSK + bcol] = p0;
    *(uint4*)&Bs[0][brow * LDSK + bcol + 8] = p1;
  }
  wait_asynccnt0();
  __syncthreads();

  int cur = 0;
  for (int k0 = 0; k0 < K; k0 += TK) {
    const int nxt = cur ^ 1;
    const bool more = (k0 + TK) < K;
    float4 f0, f1, f2, f3;
    if (more) {
      // issue next A tile to the other LDS buffer (copy engine runs during compute)
      async_load_b128(aL0[nxt], Ag0 + k0 + TK);
      async_load_b128(aL1[nxt], Ag1 + k0 + TK);
      // issue next W global loads now; consumed after the WMMA block
      const float4* ws = (const float4*)(Wg + k0 + TK);
      f0 = ws[0]; f1 = ws[1]; f2 = ws[2]; f3 = ws[3];
    }

    // ---- compute from buffer `cur` ----
    const unsigned short* Ab = &As[cur][0];
    const unsigned short* Bb = &Bs[cur][0];
    v16bf bfr[2];
#pragma unroll
    for (int fn = 0; fn < 2; ++fn) {
      int n = wn + fn * 16 + lr;
      uint4* bp = (uint4*)&bfr[fn];
      bp[0] = *(const uint4*)&Bb[n * LDSK + kB];
      bp[1] = *(const uint4*)&Bb[n * LDSK + kB + 16];
    }
#pragma unroll
    for (int fm = 0; fm < 4; ++fm) {
      v16bf afr;
      int m = wm + fm * 16 + lr;
      uint4* ap = (uint4*)&afr;
      ap[0] = *(const uint4*)&Ab[m * LDSK + kA];
      ap[1] = *(const uint4*)&Ab[m * LDSK + kA + 16];
#pragma unroll
      for (int fn = 0; fn < 2; ++fn)
        acc[fm][fn] = __builtin_amdgcn_wmma_f32_16x16x32_bf16(
            false, afr, false, bfr[fn], (short)0, acc[fm][fn], false, false);
    }

    if (more) {
      uint4 p0, p1;
      p0.x = pack2bf(f0.x, f0.y); p0.y = pack2bf(f0.z, f0.w);
      p0.z = pack2bf(f1.x, f1.y); p0.w = pack2bf(f1.z, f1.w);
      p1.x = pack2bf(f2.x, f2.y); p1.y = pack2bf(f2.z, f2.w);
      p1.z = pack2bf(f3.x, f3.y); p1.w = pack2bf(f3.z, f3.w);
      *(uint4*)&Bs[nxt][brow * LDSK + bcol] = p0;
      *(uint4*)&Bs[nxt][brow * LDSK + bcol + 8] = p1;
      wait_asynccnt0();                 // this wave's async copies into As[nxt] done
    }
    __syncthreads();                    // everyone done filling `nxt` / reading `cur`
    cur = nxt;
  }

  // epilogue (C/D layout: vgpr r -> M = r + (lane<16 ? 0 : 8), N = lane&15)
  const int rowadd = (lane < 16) ? 0 : 8;
#pragma unroll
  for (int fm = 0; fm < 4; ++fm)
#pragma unroll
    for (int fn = 0; fn < 2; ++fn) {
      int n = n0 + wn + fn * 16 + lr;
      float bsv = bias[n];
#pragma unroll
      for (int r = 0; r < 8; ++r) {
        int m = m0 + wm + fm * 16 + rowadd + r;
        float vv = acc[fm][fn][r] + bsv;
        if constexpr (MODE == 1)
          vv = 0.5f * vv * (1.0f + erff(vv * 0.70710678118654752f));
        if constexpr (MODE == 2)
          Of[(size_t)m * N + n] += vv;
        else
          Obf[(size_t)m * N + n] = f2bf(vv);
      }
    }
}

// ---------------- attention: one block per (b, head) ----------------
__global__ __launch_bounds__(128)
void attn_kernel(const unsigned short* __restrict__ Q, const unsigned short* __restrict__ Kt,
                 const unsigned short* __restrict__ V, float* __restrict__ Out)
{
  __shared__ __align__(16) float Sc[64 * 64];            // scores (f32)
  __shared__ __align__(16) unsigned short Ps[64 * 64];   // probs (bf16)
  const int b = blockIdx.x >> 1, h = blockIdx.x & 1;
  const int tid = threadIdx.x;
  const int wave = tid >> 5, lane = tid & 31;
  const int lr = lane & 15;
  const int i0 = wave * 16;                              // wave owns i-rows [i0, i0+16)
  const size_t base = (size_t)(b * S_) * D_ + (size_t)h * HD_;

  // --- scores = Q @ K^T via WMMA over K=HD ---
  v8f zero = {0.f, 0.f, 0.f, 0.f, 0.f, 0.f, 0.f, 0.f};
  v8f acc[4] = {zero, zero, zero, zero};
  const int kA = (lane < 16) ? 0 : 8;
  const int kB = (lane < 16) ? 0 : 8;
  const unsigned short* qrow = Q + base + (size_t)(i0 + lr) * D_;
  for (int e0 = 0; e0 < HD_; e0 += 32) {
    v16bf afr;
    uint4* ap = (uint4*)&afr;
    ap[0] = *(const uint4*)(qrow + e0 + kA);
    ap[1] = *(const uint4*)(qrow + e0 + kA + 16);
#pragma unroll
    for (int jt = 0; jt < 4; ++jt) {
      v16bf bfr;
      uint4* bp = (uint4*)&bfr;
      const unsigned short* krow = Kt + base + (size_t)(jt * 16 + lr) * D_ + e0 + kB;
      bp[0] = *(const uint4*)(krow);
      bp[1] = *(const uint4*)(krow + 16);
      acc[jt] = __builtin_amdgcn_wmma_f32_16x16x32_bf16(
          false, afr, false, bfr, (short)0, acc[jt], false, false);
    }
  }
  const int rowadd = (lane < 16) ? 0 : 8;
#pragma unroll
  for (int jt = 0; jt < 4; ++jt)
#pragma unroll
    for (int r = 0; r < 8; ++r)
      Sc[(i0 + rowadd + r) * 64 + jt * 16 + lr] = acc[jt][r] * (1.0f / (float)HD_);
  __syncthreads();

  // --- softmax over j<50 (note: reference divides by HD, not sqrt(HD)) ---
  if (tid < 64) {
    const int i = tid;
    float mx = -1e30f;
    for (int j = 0; j < S_; ++j) mx = fmaxf(mx, Sc[i * 64 + j]);
    float sum = 0.f;
    for (int j = 0; j < S_; ++j) {
      float e = expf(Sc[i * 64 + j] - mx);
      Sc[i * 64 + j] = e; sum += e;
    }
    float inv = 1.0f / sum;
    for (int j = 0; j < S_; ++j) Ps[i * 64 + j] = f2bf(Sc[i * 64 + j] * inv);
    for (int j = S_; j < 64; ++j) Ps[i * 64 + j] = 0;
  }
  __syncthreads();

  // --- O = P @ V, accumulate into residual ---
  for (int e = tid; e < HD_; e += 128) {
    for (int ib = 0; ib < S_; ib += 10) {
      float s[10] = {0.f, 0.f, 0.f, 0.f, 0.f, 0.f, 0.f, 0.f, 0.f, 0.f};
      for (int j = 0; j < S_; ++j) {
        float vv = bf2f(V[base + (size_t)j * D_ + e]);
#pragma unroll
        for (int t = 0; t < 10; ++t) s[t] += bf2f(Ps[(ib + t) * 64 + j]) * vv;
      }
#pragma unroll
      for (int t = 0; t < 10; ++t)
        Out[base + (size_t)(ib + t) * D_ + e] += s[t];
    }
  }
}

// ---------------- classifier + softmax ----------------
__global__ __launch_bounds__(256)
void clf_kernel(const float* __restrict__ Outb, const float* __restrict__ W,
                const float* __restrict__ bs, float* __restrict__ Y)
{
  const int b = blockIdx.x;
  const float* x = Outb + (size_t)(b * S_) * D_;   // row 0 of each sequence
  __shared__ float red[256];
  __shared__ float logits[OUT_];
  for (int o = 0; o < OUT_; ++o) {
    float sum = 0.f;
    for (int d = threadIdx.x; d < D_; d += 256) sum += x[d] * W[o * D_ + d];
    red[threadIdx.x] = sum;
    __syncthreads();
    for (int off = 128; off > 0; off >>= 1) {
      if (threadIdx.x < off) red[threadIdx.x] += red[threadIdx.x + off];
      __syncthreads();
    }
    if (threadIdx.x == 0) logits[o] = red[0] + bs[o];
    __syncthreads();
  }
  if (threadIdx.x == 0) {
    float mx = -1e30f;
    for (int o = 0; o < OUT_; ++o) mx = fmaxf(mx, logits[o]);
    float e[OUT_], sum = 0.f;
    for (int o = 0; o < OUT_; ++o) { e[o] = expf(logits[o] - mx); sum += e[o]; }
    float inv = 1.0f / sum;
    for (int o = 0; o < OUT_; ++o) Y[b * OUT_ + o] = e[o] * inv;
  }
}

// ---------------- host orchestration ----------------
extern "C" void kernel_launch(void* const* d_in, const int* in_sizes, int n_in,
                              void* d_out, int out_size, void* d_ws, size_t ws_size,
                              hipStream_t stream) {
  (void)in_sizes; (void)n_in; (void)out_size; (void)ws_size;
  const float* x        = (const float*)d_in[0];
  const float* linear_w = (const float*)d_in[1];
  const float* linear_b = (const float*)d_in[2];
  const float* cls      = (const float*)d_in[3];
  const float* ln1_g    = (const float*)d_in[4];
  const float* ln1_b    = (const float*)d_in[5];
  const float* qw       = (const float*)d_in[6];
  const float* qb       = (const float*)d_in[7];
  const float* kw       = (const float*)d_in[8];
  const float* kb       = (const float*)d_in[9];
  const float* vw       = (const float*)d_in[10];
  const float* vb       = (const float*)d_in[11];
  const float* ln2_g    = (const float*)d_in[12];
  const float* ln2_b    = (const float*)d_in[13];
  const float* w1       = (const float*)d_in[14];
  const float* b1       = (const float*)d_in[15];
  const float* w2       = (const float*)d_in[16];
  const float* b2       = (const float*)d_in[17];
  const float* clf_w    = (const float*)d_in[18];
  const float* clf_b    = (const float*)d_in[19];

  // workspace layout (all offsets multiples of 256 bytes)
  char* p = (char*)d_ws;
  float* outb = (float*)p;                   p += (size_t)M_ * D_ * sizeof(float);     // 52.4 MB residual
  unsigned short* hbuf = (unsigned short*)p; p += (size_t)M_ * D_ * 2;                 // 26.2 MB ln output
  unsigned short* qbf = (unsigned short*)p;                                            // shared region:
  unsigned short* kbf = qbf + (size_t)MPAD_ * D_;                                      //  q/k/v (78.7 MB)
  unsigned short* vbf = kbf + (size_t)MPAD_ * D_;                                      //  aliased with
  unsigned short* mid = qbf;                                                           //  mid (104.9 MB)

  embed_kernel<<<dim3(M_), dim3(256), 0, stream>>>(x, linear_w, linear_b, cls, outb);

  const dim3 gD(D_ / TN, M_ / TM);     // (8, 100)
  const dim3 gF(DFF_ / TN, M_ / TM);   // (32, 100)
  for (int l = 0; l < L_; ++l) {
    ln_kernel<<<dim3(M_), dim3(128), 0, stream>>>(outb, ln1_g + l * D_, ln1_b + l * D_, hbuf);
    gemm_bf16<0><<<gD, 256, 0, stream>>>(hbuf, qw + (size_t)l * D_ * D_, qb + l * D_, qbf, nullptr, D_, D_);
    gemm_bf16<0><<<gD, 256, 0, stream>>>(hbuf, kw + (size_t)l * D_ * D_, kb + l * D_, kbf, nullptr, D_, D_);
    gemm_bf16<0><<<gD, 256, 0, stream>>>(hbuf, vw + (size_t)l * D_ * D_, vb + l * D_, vbf, nullptr, D_, D_);
    attn_kernel<<<dim3(B_ * NH_), dim3(128), 0, stream>>>(qbf, kbf, vbf, outb);
    ln_kernel<<<dim3(M_), dim3(128), 0, stream>>>(outb, ln2_g + l * D_, ln2_b + l * D_, hbuf);
    gemm_bf16<1><<<gF, 256, 0, stream>>>(hbuf, w1 + (size_t)l * DFF_ * D_, b1 + l * DFF_, mid, nullptr, DFF_, D_);
    gemm_bf16<2><<<gD, 256, 0, stream>>>(mid, w2 + (size_t)l * D_ * DFF_, b2 + l * D_, nullptr, outb, D_, DFF_);
  }

  clf_kernel<<<dim3(B_), dim3(256), 0, stream>>>(outb, clf_w, clf_b, (float*)d_out);
}
